// TimeBlock_25005299597631
// MI455X (gfx1250) — compile-verified, matrix operand order checked by reading, not run on hardware
//
#include <hip/hip_runtime.h>
#include <hip/hip_bf16.h>

typedef __attribute__((ext_vector_type(16))) __bf16 v16bf;
typedef __attribute__((ext_vector_type(8)))  __bf16 v8bf;
typedef __attribute__((ext_vector_type(8)))  float  v8f;

#define BB 8
#define NN 207
#define TT 12
#define CC 64
#define LL (NN*TT)      /* 2484 */
#define LPAD 2496       /* 208*12 == 156*16 */
#define NPAD 208
#define LTILES 156
#define MITERS 78       /* LPAD/32 */
#define LOG2E 1.4426950408889634f

// ---------------------------------------------------------------------------
// Stage 1: fused causal conv (k=3, dil=1) x2 + GLU + q/k/v projections.
// One block per (b, n): T=12 positions, 64 channels. 64 threads (2 waves).
// Outputs:
//   qT, kT : [B][LPAD][16] bf16, row = [d0..d7 | 8 zeros] (32 B, so a whole
//            WMMA A/B fragment half-wave row is ONE contiguous v16bf load)
//   vbf    : [B][64][LPAD] bf16 (m-contiguous for P*V B-fragments)
//   h_lc   : [B][LPAD][64] f32  (epilogue residual, output layout)
//   zpad   : 64 B of zeros (load source for the all-zero high-lane rows)
// ---------------------------------------------------------------------------
__global__ __launch_bounds__(64) void stage1_conv_glu_proj(
    const float* __restrict__ X,
    const float* __restrict__ wA, const float* __restrict__ bA,
    const float* __restrict__ wB, const float* __restrict__ bB,
    const float* __restrict__ wq, const float* __restrict__ bq,
    const float* __restrict__ wk, const float* __restrict__ bk,
    const float* __restrict__ wv, const float* __restrict__ bv,
    __bf16* __restrict__ qT, __bf16* __restrict__ kT,
    __bf16* __restrict__ vbf, float* __restrict__ h_lc,
    __bf16* __restrict__ zpad)
{
  __shared__ float xs[TT][CC];
  __shared__ float hs[TT][CC];
  const int tid   = threadIdx.x;
  const int b     = blockIdx.x / NPAD;
  const int n     = blockIdx.x % NPAD;
  const int lbase = n * TT;

  if (blockIdx.x == 0 && tid < 32) zpad[tid] = (__bf16)0.f;

  if (n >= NN) {
    // zero the padded rows so stage-2 sees q=k=v=0 there
    for (int i = tid; i < TT * 16; i += 64) {
      qT[((size_t)b * LPAD + lbase) * 16 + i] = (__bf16)0.f;
      kT[((size_t)b * LPAD + lbase) * 16 + i] = (__bf16)0.f;
    }
    for (int i = tid; i < CC * TT; i += 64) {
      int c = i / TT, t = i % TT;
      vbf[((size_t)b * CC + c) * LPAD + lbase + t] = (__bf16)0.f;
    }
    for (int i = tid; i < TT * CC; i += 64)
      h_lc[((size_t)b * LPAD + lbase) * CC + i] = 0.f;
    return;
  }

  // load X[b][n][0..11][0..63] into LDS
  #pragma unroll
  for (int t = 0; t < TT; ++t)
    xs[t][tid] = X[(((size_t)b * NN + n) * TT + t) * CC + tid];
  __syncthreads();

  const int c = tid;
  float accA[TT], accB[TT];
  {
    float ba = bA[c], bb = bB[c];
    #pragma unroll
    for (int t = 0; t < TT; ++t) { accA[t] = ba; accB[t] = bb; }
  }
  // causal conv along T: y[t] = b + sum_{j,cin} w[c][cin][j] * x[t+j-2][cin]
  for (int cin = 0; cin < CC; ++cin) {
    #pragma unroll
    for (int j = 0; j < 3; ++j) {
      float wa = wA[(c * CC + cin) * 3 + j];
      float wb = wB[(c * CC + cin) * 3 + j];
      #pragma unroll
      for (int t = 0; t < TT; ++t) {
        int tt = t + j - 2;
        if (tt >= 0) { float xv = xs[tt][cin]; accA[t] += wa * xv; accB[t] += wb * xv; }
      }
    }
  }
  #pragma unroll
  for (int t = 0; t < TT; ++t)
    hs[t][c] = accA[t] * (1.0f / (1.0f + __expf(-accB[t])));  // A * sigmoid(B)
  __syncthreads();

  // v projection (out channel = c, all 12 t) + residual store
  float vacc[TT];
  {
    float bvv = bv[c];
    #pragma unroll
    for (int t = 0; t < TT; ++t) vacc[t] = bvv;
  }
  for (int cin = 0; cin < CC; ++cin) {
    float w = wv[c * CC + cin];
    #pragma unroll
    for (int t = 0; t < TT; ++t) vacc[t] += w * hs[t][cin];
  }
  #pragma unroll
  for (int t = 0; t < TT; ++t) {
    int l = lbase + t;
    vbf[((size_t)b * CC + c) * LPAD + l]   = (__bf16)vacc[t];
    h_lc[((size_t)b * LPAD + l) * CC + c]  = hs[t][c];
  }
  // q/k projections: 8 outs x 12 t = 96 tasks; zero the upper fragment half
  for (int i = tid; i < 8 * TT; i += 64) {
    int o = i & 7, t = i >> 3;
    float aq = bq[o], ak = bk[o];
    for (int cin = 0; cin < CC; ++cin) {
      float hv = hs[t][cin];
      aq += wq[o * CC + cin] * hv;
      ak += wk[o * CC + cin] * hv;
    }
    size_t row = ((size_t)b * LPAD + lbase + t) * 16;
    qT[row + o]     = (__bf16)aq;
    kT[row + o]     = (__bf16)ak;
    qT[row + 8 + o] = (__bf16)0.f;
    kT[row + 8 + o] = (__bf16)0.f;
  }
}

// ---------------------------------------------------------------------------
// Stage 2: flash attention, one wave32 per (b, 16-query-row tile).
//   S^T tile (m x l) = WMMA(A = K-tile[16m x 32d], B = Q[32d x 16l]):
//   its C-layout coincides with the A-fragment layout of P for the P*V WMMA,
//   so exp() + f32->bf16 is the only work between the two matmuls.
//   K/Q fragments are whole contiguous 32-B per-lane loads (V: two 16-B); K/Q high
//   lanes read the zero page. Softmax runs in exp2 domain.
// ---------------------------------------------------------------------------
template <bool MASK>
__device__ __forceinline__ void flash_step(
    int mbase, const __bf16* __restrict__ kp, size_t ksub,
    const __bf16* __restrict__ vRow,
    int lane15, bool lowHalf, const v16bf& bqf,
    v8f (&O)[4], float& rowmax2, float& rowsum)
{
  // two energy sub-tiles S^T (rows = m, cols = l); kp is lane-resolved and
  // the full 32-B fragment row (values | zeros) loads in one shot
  v8f s[2];
  #pragma unroll
  for (int sub = 0; sub < 2; ++sub) {
    v16bf akf = *(const v16bf*)(kp + (size_t)sub * ksub);
    s[sub] = __builtin_amdgcn_wmma_f32_16x16x32_bf16(
        false, akf, false, bqf, (short)0,
        v8f{0.f, 0.f, 0.f, 0.f, 0.f, 0.f, 0.f, 0.f}, false, false);
  }

  if (MASK) {  // only the peeled tail: mask keys with m >= L
    #pragma unroll
    for (int sub = 0; sub < 2; ++sub)
      #pragma unroll
      for (int r = 0; r < 8; ++r) {
        int m = mbase + sub * 16 + (lowHalf ? r : 8 + r);
        if (m >= LL) s[sub][r] = -1e30f;
      }
  }

  // online softmax (exp2 domain) for this lane's query l = tid&15; the
  // partner lane (xor 16) holds the other 16 m's of the same query row
  float lmax = s[0][0];
  #pragma unroll
  for (int r = 1; r < 8; ++r) lmax = fmaxf(lmax, s[0][r]);
  #pragma unroll
  for (int r = 0; r < 8; ++r) lmax = fmaxf(lmax, s[1][r]);
  float lmax2 = lmax * LOG2E;
  lmax2 = fmaxf(lmax2, __shfl_xor(lmax2, 16, 32));

  const bool grew     = lmax2 > rowmax2;
  const float newmax2 = grew ? lmax2 : rowmax2;

  v16bf ap;  // P in A-fragment layout (free: same layout as S^T's C-tile)
  float lsum = 0.f;
  #pragma unroll
  for (int r = 0; r < 8; ++r) {
    float p0 = exp2f(fmaf(s[0][r], LOG2E, -newmax2));
    float p1 = exp2f(fmaf(s[1][r], LOG2E, -newmax2));
    lsum += p0 + p1;
    ap[r]     = (__bf16)p0;
    ap[r + 8] = (__bf16)p1;
  }
  lsum += __shfl_xor(lsum, 16, 32);

  if (__any(grew)) {  // wave-uniform: EXEC stays all-ones around the WMMAs
    float alpha = exp2f(rowmax2 - newmax2);
    rowmax2 = newmax2;
    rowsum  = rowsum * alpha + lsum;
    float alr[8];
    #pragma unroll
    for (int r = 0; r < 8; ++r) alr[r] = __shfl(alpha, lowHalf ? r : 8 + r, 32);
    #pragma unroll
    for (int ch = 0; ch < 4; ++ch)
      #pragma unroll
      for (int r = 0; r < 8; ++r) O[ch][r] *= alr[r];
  } else {
    rowsum += lsum;
  }

  // P * V^T : 4 channel chunks of 16, K = 32 keys
  const int koff = lowHalf ? 0 : 8;
  #pragma unroll
  for (int ch = 0; ch < 4; ++ch) {
    const __bf16* vp = vRow + (size_t)(ch * 16 + lane15) * LPAD + mbase;
    v8bf v0 = *(const v8bf*)(vp + koff);
    v8bf v1 = *(const v8bf*)(vp + 16 + koff);
    v16bf bvf;
    #pragma unroll
    for (int i = 0; i < 8; ++i) { bvf[i] = v0[i]; bvf[i + 8] = v1[i]; }
    O[ch] = __builtin_amdgcn_wmma_f32_16x16x32_bf16(
        false, ap, false, bvf, (short)0, O[ch], false, false);
  }
}

__global__ __launch_bounds__(32) void stage2_flash_attn(
    const __bf16* __restrict__ qT, const __bf16* __restrict__ kT,
    const __bf16* __restrict__ vbf, const float* __restrict__ h_lc,
    const float* __restrict__ gamma, float* __restrict__ out,
    const __bf16* __restrict__ zpad)
{
  const int tid      = threadIdx.x;
  const int b        = blockIdx.x / LTILES;
  const int lt       = blockIdx.x % LTILES;
  const int lane15   = tid & 15;
  const bool lowHalf = tid < 16;

  const __bf16* vRow = vbf + (size_t)b * CC * LPAD;

  // lane-resolved K pointer: low lanes walk kT rows, high lanes read zeros
  const __bf16* kp  = lowHalf ? (kT + ((size_t)b * LPAD + lane15) * 16) : zpad;
  const size_t kadv = lowHalf ? (size_t)32 * 16 : 0;  // per-iteration advance
  const size_t ksub = lowHalf ? (size_t)16 * 16 : 0;  // sub-tile offset

  // B-fragment of Q: col l = lt*16 + lane15; one contiguous 32-B load
  const __bf16* qp =
      lowHalf ? (qT + ((size_t)b * LPAD + lt * 16 + lane15) * 16) : zpad;
  const v16bf bqf = *(const v16bf*)qp;

  v8f O[4];
  #pragma unroll
  for (int ch = 0; ch < 4; ++ch)
    #pragma unroll
    for (int r = 0; r < 8; ++r) O[ch][r] = 0.f;

  float rowmax2 = -1e30f, rowsum = 0.f;

  // 77 mask-free iterations (m in [0, 2464)), unrolled x2 for load/WMMA overlap
  #pragma unroll 2
  for (int it = 0; it < MITERS - 1; ++it) {
    flash_step<false>(it * 32, kp, ksub, vRow, lane15, lowHalf, bqf,
                      O, rowmax2, rowsum);
    kp += kadv;
  }
  // peeled tail (m in [2464, 2496)) with m >= L masking
  flash_step<true>((MITERS - 1) * 32, kp, ksub, vRow, lane15, lowHalf, bqf,
                   O, rowmax2, rowsum);

  // epilogue: out[b][l][c] = gamma * O/rowsum + h
  float inv[8];
  #pragma unroll
  for (int r = 0; r < 8; ++r) {
    float rs = __shfl(rowsum, lowHalf ? r : 8 + r, 32);
    inv[r] = 1.0f / rs;
  }
  const float g = gamma[0];
  #pragma unroll
  for (int ch = 0; ch < 4; ++ch) {
    int c = ch * 16 + lane15;
    #pragma unroll
    for (int r = 0; r < 8; ++r) {
      int l = lt * 16 + (lowHalf ? r : 8 + r);
      if (l < LL) {
        out[((size_t)b * LL + l) * CC + c] =
            g * (O[ch][r] * inv[r]) + h_lc[((size_t)b * LPAD + l) * CC + c];
      }
    }
  }
}

// ---------------------------------------------------------------------------
extern "C" void kernel_launch(void* const* d_in, const int* in_sizes, int n_in,
                              void* d_out, int out_size, void* d_ws, size_t ws_size,
                              hipStream_t stream) {
  (void)in_sizes; (void)n_in; (void)out_size; (void)ws_size;
  const float* X     = (const float*)d_in[0];
  const float* wA    = (const float*)d_in[1];
  const float* bA    = (const float*)d_in[2];
  const float* wB    = (const float*)d_in[3];
  const float* bB    = (const float*)d_in[4];
  const float* wq    = (const float*)d_in[5];
  const float* bq    = (const float*)d_in[6];
  const float* wk    = (const float*)d_in[7];
  const float* bk    = (const float*)d_in[8];
  const float* wv    = (const float*)d_in[9];
  const float* bv    = (const float*)d_in[10];
  const float* gamma = (const float*)d_in[11];
  float* out = (float*)d_out;

  // workspace carve (~9.0 MB total)
  char* ws = (char*)d_ws;
  __bf16* zpad = (__bf16*)ws;  ws += 64;
  __bf16* qT   = (__bf16*)ws;  ws += (size_t)BB * LPAD * 16 * sizeof(__bf16);
  __bf16* kT   = (__bf16*)ws;  ws += (size_t)BB * LPAD * 16 * sizeof(__bf16);
  __bf16* vbf  = (__bf16*)ws;  ws += (size_t)BB * CC * LPAD * sizeof(__bf16);
  float*  h_lc = (float*)ws;

  stage1_conv_glu_proj<<<BB * NPAD, 64, 0, stream>>>(
      X, wA, bA, wB, bB, wq, bq, wk, bk, wv, bv, qT, kT, vbf, h_lc, zpad);
  stage2_flash_attn<<<BB * LTILES, 32, 0, stream>>>(
      qT, kT, vbf, h_lc, gamma, out, zpad);
}
